// SlotAttention_inSlate_backup_43568148251284
// MI455X (gfx1250) — matrix operand order repaired and checked by reading, not compiled, not run on hardware
//
#include <hip/hip_runtime.h>
#include <hip/hip_bf16.h>

typedef _Float16 h16;
typedef __attribute__((ext_vector_type(16))) _Float16 v16h;
typedef __attribute__((ext_vector_type(8)))  _Float16 v8h;
typedef __attribute__((ext_vector_type(8)))  float    v8f;

#define WMMA_F32_F16(a, b, c) \
  __builtin_amdgcn_wmma_f32_16x16x32_f16(false, (a), false, (b), (short)0, (c), false, false)

constexpr int B   = 32;
constexpr int NKV = 4096;
constexpr int DIN = 256;
constexpr int NQ  = 16;
constexpr int S   = 512;
constexpr int H   = 8;
constexpr int HD  = 64;
constexpr int HM  = 512;
constexpr int HQ  = H * NQ;       // 128
constexpr int BQS = B * NQ * S;   // 262144
constexpr int KSPLIT = 4;         // K-chunks for updates GEMM
constexpr float EPSN = 1e-8f;

// ---- WMMA fragment loaders (CDNA5 16-bit layouts, ISA 7.12.2) -------------
// A 16x32: lane holds row m=lane&15; elems 0-7 -> K=hs*8+i, elems 8-15 -> K=16+hs*8+i
__device__ __forceinline__ v16h frag_a(const h16* rb, int hs) {
  v8h lo = *(const v8h*)(rb + hs * 8);
  v8h hi = *(const v8h*)(rb + 16 + hs * 8);
  v16h r;
#pragma unroll
  for (int i = 0; i < 8; ++i) { r[i] = lo[i]; r[i + 8] = hi[i]; }
  return r;
}
// B 32x16: lane holds col n=lane&15; elems 0-7 -> K=hs*8+i, elems 8-15 -> K=16+hs*8+i
__device__ __forceinline__ v16h frag_b(const h16* rb, int hs) {
  v8h lo = *(const v8h*)(rb + hs * 8);
  v8h hi = *(const v8h*)(rb + 16 + hs * 8);
  v16h r;
#pragma unroll
  for (int i = 0; i < 8; ++i) { r[i] = lo[i]; r[i + 8] = hi[i]; }
  return r;
}

// ---- small utility kernels -------------------------------------------------
__global__ void cvt_f32_f16(const float* __restrict__ in, h16* __restrict__ out, int n) {
  int i = blockIdx.x * blockDim.x + threadIdx.x;
  if (i < n) out[i] = (h16)in[i];
}

__global__ void copy_f32(const float* __restrict__ in, float* __restrict__ out, int n) {
  int i = blockIdx.x * blockDim.x + threadIdx.x;
  if (i < n) out[i] = in[i];
}

__global__ void init_slots(const float* __restrict__ sl, float* __restrict__ out) {
  int i = blockIdx.x * blockDim.x + threadIdx.x;
  if (i < BQS) out[i] = sl[i % (NQ * S)];
}

// LayerNorm over rows (width 256 or 512); optional raw f16 copy of the input row.
__global__ void ln_rows(const float* __restrict__ in, const float* __restrict__ g,
                        const float* __restrict__ bta, h16* __restrict__ out_ln,
                        h16* __restrict__ out_raw, int rows, int width) {
  int wid  = (blockIdx.x * blockDim.x + threadIdx.x) >> 5;
  int lane = threadIdx.x & 31;
  if (wid >= rows) return;
  const float* row = in + (size_t)wid * width;
  int per = width >> 5;                 // 8 or 16
  float v[16];
  float s = 0.f;
  for (int i = 0; i < per; ++i) { v[i] = row[lane + 32 * i]; s += v[i]; }
  for (int o = 16; o > 0; o >>= 1) s += __shfl_xor(s, o, 32);
  float mu = s / width;
  float s2 = 0.f;
  for (int i = 0; i < per; ++i) { float d = v[i] - mu; s2 += d * d; }
  for (int o = 16; o > 0; o >>= 1) s2 += __shfl_xor(s2, o, 32);
  float rs = rsqrtf(s2 / width + 1e-5f);
  for (int i = 0; i < per; ++i) {
    int c = lane + 32 * i;
    out_ln[(size_t)wid * width + c] = (h16)((v[i] - mu) * rs * g[c] + bta[c]);
    if (out_raw) out_raw[(size_t)wid * width + c] = (h16)v[i];
  }
}

// ---- generic C = A @ W.T (+bias)(+relu)(+add); 2x2 register-blocked -------
enum { GF_BIAS = 1, GF_RELU = 2, GF_ADD = 4, GF_OUTF = 8, GF_OUTH = 16 };

__global__ void gemm_awt(const h16* __restrict__ A, const h16* __restrict__ W,
                         const float* __restrict__ bias, const float* __restrict__ add,
                         float* __restrict__ outF, h16* __restrict__ outH,
                         int M, int N, int Kd, int flags) {
  int wid  = (blockIdx.x * blockDim.x + threadIdx.x) >> 5;
  int lane = threadIdx.x & 31;
  int tn = N >> 5;                      // 32-wide col blocks
  int tiles = (M >> 5) * tn;
  if (wid >= tiles) return;
  int tm = wid / tn, tc = wid % tn;
  int hs = lane >> 4, l15 = lane & 15;
  const h16* a0 = A + (size_t)(tm * 32 + l15) * Kd;
  const h16* a1 = a0 + (size_t)16 * Kd;
  const h16* b0 = W + (size_t)(tc * 32 + l15) * Kd;
  const h16* b1 = b0 + (size_t)16 * Kd;
  v8f cc[2][2] = {};
  for (int k0 = 0; k0 < Kd; k0 += 32) {
    v16h fa0 = frag_a(a0 + k0, hs), fa1 = frag_a(a1 + k0, hs);
    v16h fb0 = frag_b(b0 + k0, hs), fb1 = frag_b(b1 + k0, hs);
    cc[0][0] = WMMA_F32_F16(fa0, fb0, cc[0][0]);
    cc[0][1] = WMMA_F32_F16(fa0, fb1, cc[0][1]);
    cc[1][0] = WMMA_F32_F16(fa1, fb0, cc[1][0]);
    cc[1][1] = WMMA_F32_F16(fa1, fb1, cc[1][1]);
  }
#pragma unroll
  for (int i = 0; i < 2; ++i)
#pragma unroll
    for (int j = 0; j < 2; ++j) {
      int n = tc * 32 + j * 16 + l15;
#pragma unroll
      for (int r = 0; r < 8; ++r) {
        int m = tm * 32 + i * 16 + r + 8 * hs;
        size_t o = (size_t)m * N + n;
        float v = cc[i][j][r];
        if (flags & GF_BIAS) v += bias[n];
        if (flags & GF_RELU) v = fmaxf(v, 0.f);
        if (flags & GF_ADD)  v += add[o];
        if (flags & GF_OUTF) outF[o] = v;
        if (flags & GF_OUTH) outH[o] = (h16)v;
      }
    }
}

// ---- K/V projection (2x2 blocked): K scaled by hd^-0.5, V -> [B,S,NKV] ----
__global__ void kv_gemm(const h16* __restrict__ X, const h16* __restrict__ Wkv,
                        h16* __restrict__ Kh, h16* __restrict__ Vt) {
  int wid  = (blockIdx.x * blockDim.x + threadIdx.x) >> 5;
  int lane = threadIdx.x & 31;
  const int TN = (2 * S) >> 5;                 // 32 col blocks; 0..15 = K, 16..31 = V
  int tiles = ((B * NKV) >> 5) * TN;
  if (wid >= tiles) return;
  int tm = wid / TN, tc = wid % TN;
  int hs = lane >> 4, l15 = lane & 15;
  const h16* a0 = X + (size_t)(tm * 32 + l15) * DIN;
  const h16* a1 = a0 + (size_t)16 * DIN;
  const h16* b0 = Wkv + (size_t)(tc * 32 + l15) * DIN;
  const h16* b1 = b0 + (size_t)16 * DIN;
  v8f cc[2][2] = {};
  for (int k0 = 0; k0 < DIN; k0 += 32) {
    v16h fa0 = frag_a(a0 + k0, hs), fa1 = frag_a(a1 + k0, hs);
    v16h fb0 = frag_b(b0 + k0, hs), fb1 = frag_b(b1 + k0, hs);
    cc[0][0] = WMMA_F32_F16(fa0, fb0, cc[0][0]);
    cc[0][1] = WMMA_F32_F16(fa0, fb1, cc[0][1]);
    cc[1][0] = WMMA_F32_F16(fa1, fb0, cc[1][0]);
    cc[1][1] = WMMA_F32_F16(fa1, fb1, cc[1][1]);
  }
#pragma unroll
  for (int i = 0; i < 2; ++i)
#pragma unroll
    for (int j = 0; j < 2; ++j) {
      int n = tc * 32 + j * 16 + l15;
#pragma unroll
      for (int r = 0; r < 8; ++r) {
        int m = tm * 32 + i * 16 + r + 8 * hs;   // global token row
        if (tc < 16) {
          Kh[(size_t)m * S + n] = (h16)(cc[i][j][r] * 0.125f);  // hd^-0.5 folded in
        } else {
          int b = m >> 12, kk = m & (NKV - 1);
          Vt[((size_t)b * S + (n - S)) * NKV + kk] = (h16)cc[i][j][r];
        }
      }
    }
}

// ---- fused logits (per-head WMMA) + joint softmax(128) + attnT/dpart/vis --
__global__ void attn_fused(const h16* __restrict__ Kh, const h16* __restrict__ Qh,
                           h16* __restrict__ attnT, float* __restrict__ dpart,
                           float* __restrict__ vis) {
  __shared__ float lg[16][HQ];                  // 16 keys x 128 (h,q)
  int b  = blockIdx.x >> 8;                     // 256 key-tiles per batch
  int kt = blockIdx.x & 255;
  int w  = threadIdx.x >> 5;                    // wave id == head
  int lane = threadIdx.x & 31;
  int hs = lane >> 4, l15 = lane & 15;
  const h16* arow = Kh + (size_t)(b * NKV + kt * 16 + l15) * S + w * HD;
  const h16* brow = Qh + (size_t)(b * NQ + l15) * S + w * HD;
  v8f acc = {};
  acc = WMMA_F32_F16(frag_a(arow, hs),      frag_b(brow, hs),      acc);
  acc = WMMA_F32_F16(frag_a(arow + 32, hs), frag_b(brow + 32, hs), acc);
#pragma unroll
  for (int r = 0; r < 8; ++r) lg[r + 8 * hs][w * 16 + l15] = acc[r];
  __syncthreads();

  // softmax over 128 per key row: 16 threads/row, 8 cols each (stride 16)
  int rrow = threadIdx.x >> 4;
  int c16  = threadIdx.x & 15;
  float vals[8], mx = -3.4e38f;
#pragma unroll
  for (int j = 0; j < 8; ++j) { vals[j] = lg[rrow][c16 + 16 * j]; mx = fmaxf(mx, vals[j]); }
  for (int o = 8; o > 0; o >>= 1) mx = fmaxf(mx, __shfl_xor(mx, o, 32));
  float sum = 0.f;
#pragma unroll
  for (int j = 0; j < 8; ++j) { vals[j] = expf(vals[j] - mx); sum += vals[j]; }
  for (int o = 8; o > 0; o >>= 1) sum += __shfl_xor(sum, o, 32);
  float inv = 1.0f / sum;
  __syncthreads();
#pragma unroll
  for (int j = 0; j < 8; ++j) {
    float a = vals[j] * inv;
    int c = c16 + 16 * j;
    lg[rrow][c] = a;
    attnT[((size_t)b * HQ + c) * NKV + kt * 16 + rrow] = (h16)a;
  }
  __syncthreads();
  if (threadIdx.x < HQ) {                       // deterministic key-axis partials
    float sp = 0.f;
#pragma unroll
    for (int r = 0; r < 16; ++r) sp += lg[r][threadIdx.x];
    dpart[((size_t)b * 256 + kt) * HQ + threadIdx.x] = sp;
  }
  {                                             // attn_vis = sum over heads
    int r = threadIdx.x >> 4, q = threadIdx.x & 15;
    float sp = 0.f;
#pragma unroll
    for (int hh = 0; hh < H; ++hh) sp += lg[r][hh * 16 + q];
    vis[((size_t)b * NKV + kt * 16 + r) * NQ + q] = sp;
  }
}

__global__ void denom_reduce(const float* __restrict__ dpart, float* __restrict__ denom) {
  int i = blockIdx.x * blockDim.x + threadIdx.x;     // B*HQ = 4096
  if (i >= B * HQ) return;
  int b = i >> 7, c = i & 127;
  float s = 0.f;
  for (int t = 0; t < 256; ++t) s += dpart[((size_t)b * 256 + t) * HQ + c];
  denom[i] = s;
}

// ---- updates: U[q,d] = attnT[q,:] @ Vt[d,:] per (b,h); K split, 1x2 cols --
__global__ void updates_gemm(const h16* __restrict__ attnT, const h16* __restrict__ Vt,
                             float* __restrict__ updp /*[KSPLIT][B*NQ*S]*/) {
  int wid  = (blockIdx.x * blockDim.x + threadIdx.x) >> 5;   // 2048 waves
  int lane = threadIdx.x & 31;
  int kc  = wid >> 9;            // K chunk 0..3
  int rem = wid & 511;
  int b = rem >> 4; int r2 = rem & 15; int hh = r2 >> 1; int dt = r2 & 1;
  int hs = lane >> 4, l15 = lane & 15;
  const int KC = NKV / KSPLIT;   // 1024
  const h16* arow = attnT + ((size_t)b * HQ + hh * 16 + l15) * NKV + kc * KC;
  const h16* v0 = Vt + ((size_t)b * S + hh * HD + dt * 32 + l15) * NKV + kc * KC;
  const h16* v1 = v0 + (size_t)16 * NKV;
  v8f c0 = {}, c1 = {};
  for (int k0 = 0; k0 < KC; k0 += 32) {
    v16h fa = frag_a(arow + k0, hs);
    c0 = WMMA_F32_F16(fa, frag_b(v0 + k0, hs), c0);
    c1 = WMMA_F32_F16(fa, frag_b(v1 + k0, hs), c1);
  }
  float* out = updp + (size_t)kc * BQS;
#pragma unroll
  for (int j = 0; j < 2; ++j) {
    v8f acc = j ? c1 : c0;
    int col = hh * HD + dt * 32 + j * 16 + l15;
#pragma unroll
    for (int r = 0; r < 8; ++r) {
      int q = r + 8 * hs;
      out[((size_t)(b * NQ + q)) * S + col] = acc[r];
    }
  }
}

__global__ void updates_finalize(const float* __restrict__ updp,
                                 const float* __restrict__ denom,
                                 h16* __restrict__ upd) {
  int i = blockIdx.x * blockDim.x + threadIdx.x;
  if (i >= BQS) return;
  int row = i >> 9, col = i & 511;
  int b = row >> 4, q = row & 15, hh = col >> 6;
  float s = updp[i] + updp[i + BQS] + updp[i + 2 * BQS] + updp[i + 3 * BQS];
  upd[i] = (h16)(s / (denom[b * HQ + hh * 16 + q] + EPSN));
}

// ---- GRU cell elementwise --------------------------------------------------
__global__ void gru_cell(const float* __restrict__ gi, const float* __restrict__ gh,
                         float* __restrict__ slots) {
  int i = blockIdx.x * blockDim.x + threadIdx.x;
  if (i >= BQS) return;
  int row = i >> 9, col = i & 511;
  size_t base = (size_t)row * 3 * S + col;
  float ir = gi[base], iz = gi[base + S], inn = gi[base + 2 * S];
  float hr = gh[base], hz = gh[base + S], hn  = gh[base + 2 * S];
  float h = slots[i];
  float r = 1.f / (1.f + expf(-(ir + hr)));
  float z = 1.f / (1.f + expf(-(iz + hz)));
  float n = tanhf(inn + r * hn);
  slots[i] = (1.f - z) * n + z * h;
}

extern "C" void kernel_launch(void* const* d_in, const int* in_sizes, int n_in,
                              void* d_out, int out_size, void* d_ws, size_t ws_size,
                              hipStream_t stream) {
  (void)in_sizes; (void)n_in; (void)out_size; (void)ws_size;
  const float* slots_in = (const float*)d_in[0];
  const float* inputs   = (const float*)d_in[1];
  const float* ning = (const float*)d_in[2];
  const float* ninb = (const float*)d_in[3];
  const float* nsg  = (const float*)d_in[4];
  const float* nsb  = (const float*)d_in[5];
  const float* nmg  = (const float*)d_in[6];
  const float* nmb  = (const float*)d_in[7];
  const float* Wq   = (const float*)d_in[8];
  const float* Wk   = (const float*)d_in[9];
  const float* Wv   = (const float*)d_in[10];
  const float* wih  = (const float*)d_in[11];
  const float* whh  = (const float*)d_in[12];
  const float* bih  = (const float*)d_in[13];
  const float* bhh  = (const float*)d_in[14];
  const float* w1   = (const float*)d_in[15];
  const float* b1   = (const float*)d_in[16];
  const float* w2   = (const float*)d_in[17];
  const float* b2   = (const float*)d_in[18];

  float* out_slots = (float*)d_out;
  float* out_vis   = out_slots + (size_t)BQS;

  char* p = (char*)d_ws;
  auto alloc = [&](size_t bytes) -> char* {
    char* r = p; p += (bytes + 255) & ~size_t(255); return r;
  };
  h16*   xh    = (h16*)  alloc((size_t)B * NKV * DIN * 2);
  h16*   Kh    = (h16*)  alloc((size_t)B * NKV * S * 2);
  h16*   Vt    = (h16*)  alloc((size_t)B * S * NKV * 2);
  h16*   attnT = (h16*)  alloc((size_t)B * HQ * NKV * 2);
  float* slots = (float*)alloc((size_t)BQS * 4);
  h16*   sln   = (h16*)  alloc((size_t)BQS * 2);
  h16*   sraw  = (h16*)  alloc((size_t)BQS * 2);
  h16*   Qh    = (h16*)  alloc((size_t)BQS * 2);
  float* dpart = (float*)alloc((size_t)B * 256 * HQ * 4);
  float* denom = (float*)alloc((size_t)B * HQ * 4);
  float* updp  = (float*)alloc((size_t)KSPLIT * BQS * 4);
  h16*   upd   = (h16*)  alloc((size_t)BQS * 2);
  float* gi    = (float*)alloc((size_t)B * NQ * 3 * S * 4);
  float* gh    = (float*)alloc((size_t)B * NQ * 3 * S * 4);
  h16*   mh    = (h16*)  alloc((size_t)BQS * 2);
  h16*   h1    = (h16*)  alloc((size_t)B * NQ * HM * 2);
  h16*   Wqh   = (h16*)  alloc((size_t)S * S * 2);
  h16*   Wkvh  = (h16*)  alloc((size_t)2 * S * DIN * 2);
  h16*   wihh  = (h16*)  alloc((size_t)3 * S * S * 2);
  h16*   whhh  = (h16*)  alloc((size_t)3 * S * S * 2);
  h16*   w1h   = (h16*)  alloc((size_t)HM * S * 2);
  h16*   w2h   = (h16*)  alloc((size_t)S * HM * 2);

  auto cvt = [&](const float* src, h16* dst, int n) {
    cvt_f32_f16<<<(n + 255) / 256, 256, 0, stream>>>(src, dst, n);
  };
  cvt(Wq, Wqh, S * S);
  cvt(Wk, Wkvh, S * DIN);
  cvt(Wv, Wkvh + (size_t)S * DIN, S * DIN);
  cvt(wih, wihh, 3 * S * S);
  cvt(whh, whhh, 3 * S * S);
  cvt(w1, w1h, HM * S);
  cvt(w2, w2h, S * HM);

  { int rows = B * NKV;
    ln_rows<<<(rows * 32 + 255) / 256, 256, 0, stream>>>(inputs, ning, ninb, xh, nullptr, rows, DIN); }
  { int tiles = ((B * NKV) / 32) * ((2 * S) / 32);
    kv_gemm<<<tiles / 8, 256, 0, stream>>>(xh, Wkvh, Kh, Vt); }
  init_slots<<<(BQS + 255) / 256, 256, 0, stream>>>(slots_in, slots);

  for (int it = 0; it < 3; ++it) {
    int rows = B * NQ;  // 512
    ln_rows<<<(rows * 32 + 255) / 256, 256, 0, stream>>>(slots, nsg, nsb, sln, sraw, rows, S);
    { int tiles = (rows / 32) * (S / 32);
      gemm_awt<<<tiles / 8, 256, 0, stream>>>(sln, Wqh, nullptr, nullptr, nullptr, Qh,
                                              rows, S, S, GF_OUTH); }
    attn_fused<<<B * 256, 256, 0, stream>>>(Kh, Qh, attnT, dpart, out_vis);
    denom_reduce<<<(B * HQ + 255) / 256, 256, 0, stream>>>(dpart, denom);
    updates_gemm<<<(2048 * 32) / 256, 256, 0, stream>>>(attnT, Vt, updp);
    updates_finalize<<<(BQS + 255) / 256, 256, 0, stream>>>(updp, denom, upd);
    { int tiles = (rows / 32) * ((3 * S) / 32);
      gemm_awt<<<tiles / 8, 256, 0, stream>>>(upd, wihh, bih, nullptr, gi, nullptr,
                                              rows, 3 * S, S, GF_BIAS | GF_OUTF);
      gemm_awt<<<tiles / 8, 256, 0, stream>>>(sraw, whhh, bhh, nullptr, gh, nullptr,
                                              rows, 3 * S, S, GF_BIAS | GF_OUTF); }
    gru_cell<<<(BQS + 255) / 256, 256, 0, stream>>>(gi, gh, slots);
    ln_rows<<<(rows * 32 + 255) / 256, 256, 0, stream>>>(slots, nmg, nmb, mh, nullptr, rows, S);
    { int tiles = (rows / 32) * (HM / 32);
      gemm_awt<<<tiles / 8, 256, 0, stream>>>(mh, w1h, b1, nullptr, nullptr, h1,
                                              rows, HM, S, GF_BIAS | GF_RELU | GF_OUTH); }
    { int tiles = (rows / 32) * (S / 32);
      gemm_awt<<<tiles / 8, 256, 0, stream>>>(h1, w2h, b2, slots, slots, nullptr,
                                              rows, S, HM, GF_BIAS | GF_ADD | GF_OUTF); }
  }
  copy_f32<<<(BQS + 255) / 256, 256, 0, stream>>>(slots, out_slots, BQS);
}
